// DecAttention_5531917877896
// MI455X (gfx1250) — matrix operand order repaired and checked by reading, not compiled, not run on hardware
//
#include <hip/hip_runtime.h>
#include <hip/hip_bf16.h>

// ---------- types ----------
typedef __attribute__((ext_vector_type(16))) __bf16 bf16x16;
typedef __attribute__((ext_vector_type(8)))  float  f32x8;

union Frag {
  bf16x16        v;
  unsigned int   u[8];
  unsigned short h[16];
  uint4          q[2];
};

#define WMMA_BF16(a, b, c) \
  __builtin_amdgcn_wmma_f32_16x16x32_bf16(false, (a), false, (b), (short)0, (c), false, false)

__device__ __forceinline__ unsigned short f2bf(float f) {
  unsigned int u = __float_as_uint(f);
  unsigned int r = u + 0x7FFFu + ((u >> 16) & 1u);   // round-to-nearest-even
  return (unsigned short)(r >> 16);
}
__device__ __forceinline__ unsigned int pk2bf(float lo, float hi) {
  return (unsigned int)f2bf(lo) | ((unsigned int)f2bf(hi) << 16);
}

// Problem constants: B=4, T=2048, C=1024, H=16, D=64

// =====================================================================
// Kernel 1: qkv = x @ Wqkv + bqkv ; scatter to Q[B,H,T,D], K[B,H,T,D],
//           Vt[B,H,D,T] (all bf16).  M=8192, N=3072, K=1024.
// Block: 256 thr (8 waves), tile 64(M) x 256(N), K-step 32.
// Wave: wm=w&1 (32 rows), wn=w>>1 (64 cols) -> 2 A-frag x 4 B-frag
// = 8 wmma/step with register reuse.
// B staged pair-row packed: Bs2[k/2][n] = {bf16(k even), bf16(k odd)}.
// =====================================================================
__global__ __launch_bounds__(256) void qkv_gemm_kernel(
    const float* __restrict__ x, const float* __restrict__ Wqkv,
    const float* __restrict__ bqkv,
    unsigned short* __restrict__ Qo, unsigned short* __restrict__ Ko,
    unsigned short* __restrict__ Vt)
{
  __shared__ unsigned short As[64][40];      // [m][k], 16B-aligned rows
  __shared__ unsigned int   Bs2[16][264];    // [k/2][n] packed pairs

  const int tid   = threadIdx.x;
  const int lane  = tid & 31;
  const int w     = tid >> 5;
  const int wm    = w & 1;
  const int wn    = w >> 1;
  const int mblk  = blockIdx.x * 64;
  const int nblk  = blockIdx.y * 256;
  const int nhalf = lane >> 4;
  const int n16   = lane & 15;

  f32x8 acc[2][4];
  #pragma unroll
  for (int bn = 0; bn < 4; ++bn) {
    float bv = bqkv[nblk + wn * 64 + bn * 16 + n16];
    #pragma unroll
    for (int am = 0; am < 2; ++am)
      #pragma unroll
      for (int r = 0; r < 8; ++r) acc[am][bn][r] = bv;
  }

  for (int kk = 0; kk < 1024; kk += 32) {
    __syncthreads();
    // stage A: x[mblk..+63][kk..+31] -> bf16, float4 granularity
    #pragma unroll
    for (int i = 0; i < 2; ++i) {
      int idx = tid + i * 256;                 // 512 float4 chunks
      int row = idx >> 3, c4 = idx & 7;
      float4 f = *(const float4*)(x + (size_t)(mblk + row) * 1024 + kk + c4 * 4);
      uint2 p;
      p.x = pk2bf(f.x, f.y);
      p.y = pk2bf(f.z, f.w);
      *(uint2*)&As[row][c4 * 4] = p;
    }
    // stage B pair-packed: rows kk+2p,kk+2p+1 of Wqkv, cols nblk..+255
    #pragma unroll
    for (int i = 0; i < 4; ++i) {
      int idx = tid + i * 256;                 // 1024 uint4 chunks
      int pr = idx >> 6, ch = idx & 63;
      const float* src = Wqkv + (size_t)(kk + 2 * pr) * 3072 + nblk + ch * 4;
      float4 r0 = *(const float4*)(src);
      float4 r1 = *(const float4*)(src + 3072);
      uint4 p;
      p.x = pk2bf(r0.x, r1.x);
      p.y = pk2bf(r0.y, r1.y);
      p.z = pk2bf(r0.z, r1.z);
      p.w = pk2bf(r0.w, r1.w);
      *(uint4*)&Bs2[pr][ch * 4] = p;
    }
    if (kk + 32 < 1024) {
      __builtin_prefetch(x + (size_t)(mblk + (tid & 63)) * 1024 + kk + 32, 0, 0);
      __builtin_prefetch(Wqkv + (size_t)(kk + 32 + (tid & 31)) * 3072 + nblk, 0, 0);
    }
    __syncthreads();

    // A fragments: 16x32, row = lane&15; v0..3: K=khalf+2v, v4..7: K=16+khalf+2v
    Frag a[2], b[4];
    #pragma unroll
    for (int am = 0; am < 2; ++am) {
      int m = wm * 32 + am * 16 + n16;
      int khalf = nhalf * 8;
      #pragma unroll
      for (int v = 0; v < 4; ++v)
        a[am].u[v] = *(const unsigned int*)&As[m][khalf + 2 * v];
      #pragma unroll
      for (int v = 0; v < 4; ++v)
        a[am].u[4 + v] = *(const unsigned int*)&As[m][16 + khalf + 2 * v];
    }
    // B fragments: 32x16, N=lane&15, v0..3: K=nhalf*8+2v, v4..7: K=16+nhalf*8+2(v-4) (pre-packed)
    #pragma unroll
    for (int bn = 0; bn < 4; ++bn) {
      int n = wn * 64 + bn * 16 + n16;
      int kh2 = nhalf * 4;
      #pragma unroll
      for (int v = 0; v < 8; ++v)
        b[bn].u[v] = Bs2[(v >> 2) * 8 + kh2 + (v & 3)][n];
    }
    #pragma unroll
    for (int am = 0; am < 2; ++am)
      #pragma unroll
      for (int bn = 0; bn < 4; ++bn)
        acc[am][bn] = WMMA_BF16(a[am].v, b[bn].v, acc[am][bn]);
  }

  // Epilogue: part/head/d-base wave-uniform per bn -> scalar branches.
  #pragma unroll
  for (int bn = 0; bn < 4; ++bn) {
    int gn0  = __builtin_amdgcn_readfirstlane(nblk + wn * 64 + bn * 16);
    int part = gn0 >> 10;          // 0:Q 1:K 2:V
    int hh   = (gn0 & 1023) >> 6;
    int d0   = gn0 & 63;
    #pragma unroll
    for (int am = 0; am < 2; ++am) {
      #pragma unroll
      for (int r = 0; r < 8; ++r) {
        int gm = mblk + wm * 32 + am * 16 + r + 8 * nhalf;
        int bb = gm >> 11, tt = gm & 2047;
        size_t bhI = (size_t)bb * 16 + hh;
        unsigned short hv = f2bf(acc[am][bn][r]);
        if (part == 0)      Qo[(bhI * 2048 + tt) * 64 + d0 + n16] = hv;
        else if (part == 1) Ko[(bhI * 2048 + tt) * 64 + d0 + n16] = hv;
        else                Vt[(bhI * 64 + d0 + n16) * 2048 + tt] = hv;
      }
    }
  }
}

// =====================================================================
// Kernel 2: causal flash attention. One wave per 16-query tile.
// 8192 waves -> 2048 blocks x 128 threads. Key step = 32.
// Per step: 4 wmma for S (=Q K^T over D=64), online softmax (co-executes
// with XDL wmma), P via per-wave LDS layout fixup, 4 wmma for O += P V.
// =====================================================================
__global__ __launch_bounds__(128) void attn_kernel(
    const unsigned short* __restrict__ Q,
    const unsigned short* __restrict__ K,
    const unsigned short* __restrict__ Vt,
    unsigned short* __restrict__ Y)
{
  __shared__ unsigned short Plds[4][16][32];   // per-wave 16x32 bf16 P tile

  const int lane  = threadIdx.x & 31;
  const int wv    = threadIdx.x >> 5;
  const int wg    = blockIdx.x * 4 + wv;
  const int qt    = wg & 127;           // query tile (T/16 = 128)
  const int bh    = wg >> 7;            // 0..63  (= b*16 + h)
  const int qbase = qt * 16;
  const int nhalf = lane >> 4;
  const int n16   = lane & 15;
  const int mrow  = nhalf * 8;

  const size_t qkb = (size_t)bh * 2048 * 64;   // Q/K base for this head
  const size_t vb  = (size_t)bh * 64 * 2048;   // Vt base

  // Q A-fragments (d 0..31 and 32..63): row = lane&15, contiguous 16B chunks
  Frag aQ0, aQ1;
  {
    size_t base = qkb + (size_t)(qbase + n16) * 64;
    int khalf = nhalf * 8;
    aQ0.q[0] = *(const uint4*)(Q + base + khalf);
    aQ0.q[1] = *(const uint4*)(Q + base + 16 + khalf);
    aQ1.q[0] = *(const uint4*)(Q + base + 32 + khalf);
    aQ1.q[1] = *(const uint4*)(Q + base + 48 + khalf);
  }

  f32x8 o[4];
  #pragma unroll
  for (int t = 0; t < 4; ++t)
    #pragma unroll
    for (int r = 0; r < 8; ++r) o[t][r] = 0.f;
  float mrun[8], lrun[8];
  #pragma unroll
  for (int r = 0; r < 8; ++r) { mrun[r] = -3.0e38f; lrun[r] = 0.f; }

  const float SC = 0.125f;   // 1/sqrt(64)

  for (int j = 0; j < qbase + 16; j += 32) {
    // ---- S = Q K^T (two 16x16 tiles over keys j..j+31, K-dim = 64) ----
    f32x8 s[2];
    #pragma unroll
    for (int sb = 0; sb < 2; ++sb) {
      Frag bk0, bk1;   // B 32x16: N=key (lane&15), K=d split by lane half
      size_t kb = qkb + (size_t)(j + sb * 16 + n16) * 64;
      bk0.q[0] = *(const uint4*)(K + kb + nhalf * 8);
      bk0.q[1] = *(const uint4*)(K + kb + 16 + nhalf * 8);
      bk1.q[0] = *(const uint4*)(K + kb + 32 + nhalf * 8);
      bk1.q[1] = *(const uint4*)(K + kb + 48 + nhalf * 8);
      f32x8 z = {};
      z = WMMA_BF16(aQ0.v, bk0.v, z);
      s[sb] = WMMA_BF16(aQ1.v, bk1.v, z);
    }
    // ---- scale + causal mask (C layout: row = mrow+r, col = n16) ----
    #pragma unroll
    for (int sb = 0; sb < 2; ++sb) {
      int kcol = j + sb * 16 + n16;
      #pragma unroll
      for (int r = 0; r < 8; ++r) {
        int qrow = qbase + mrow + r;
        float vv = s[sb][r] * SC;
        s[sb][r] = (kcol > qrow) ? -3.0e38f : vv;
      }
    }
    // ---- row max over the 16 lanes of this half + online rescale ----
    float alpha[8];
    #pragma unroll
    for (int r = 0; r < 8; ++r) {
      float a0 = s[0][r], a1 = s[1][r];
      #pragma unroll
      for (int off = 1; off < 16; off <<= 1) {
        a0 = fmaxf(a0, __shfl_xor(a0, off, 32));
        a1 = fmaxf(a1, __shfl_xor(a1, off, 32));
      }
      float mn = fmaxf(mrun[r], fmaxf(a0, a1));
      alpha[r] = __expf(mrun[r] - mn);
      mrun[r] = mn;
    }
    // ---- P = exp(S - m), row sums ----
    float p0[8], p1[8];
    #pragma unroll
    for (int r = 0; r < 8; ++r) {
      p0[r] = __expf(s[0][r] - mrun[r]);
      p1[r] = __expf(s[1][r] - mrun[r]);
      float t0 = p0[r], t1 = p1[r];
      #pragma unroll
      for (int off = 1; off < 16; off <<= 1) {
        t0 += __shfl_xor(t0, off, 32);
        t1 += __shfl_xor(t1, off, 32);
      }
      lrun[r] = lrun[r] * alpha[r] + t0 + t1;
    }
    #pragma unroll
    for (int t = 0; t < 4; ++t)
      #pragma unroll
      for (int r = 0; r < 8; ++r) o[t][r] *= alpha[r];

    // ---- C-layout f32 -> A-layout bf16 via per-wave LDS tile ----
    #pragma unroll
    for (int r = 0; r < 8; ++r) {
      Plds[wv][mrow + r][n16]      = f2bf(p0[r]);
      Plds[wv][mrow + r][16 + n16] = f2bf(p1[r]);
    }
    __builtin_amdgcn_wave_barrier();
    asm volatile("s_wait_dscnt 0" ::: "memory");
    Frag aP;
    {
      int khalf = nhalf * 16;
      aP.q[0] = *(const uint4*)&Plds[wv][n16][khalf];
      aP.q[1] = *(const uint4*)&Plds[wv][n16][khalf + 8];
    }
    __builtin_amdgcn_wave_barrier();

    // ---- O += P V : Vt gives B 32x16 fragments as contiguous loads ----
    #pragma unroll
    for (int dt = 0; dt < 4; ++dt) {
      Frag bvf;
      size_t va = vb + (size_t)(dt * 16 + n16) * 2048 + j + nhalf * 16;
      bvf.q[0] = *(const uint4*)(Vt + va);
      bvf.q[1] = *(const uint4*)(Vt + va + 8);
      o[dt] = WMMA_BF16(aP.v, bvf.v, o[dt]);
    }
  }

  // ---- normalize and store Y[B*T][C] as bf16 ----
  const int bb = bh >> 4, hh = bh & 15;
  #pragma unroll
  for (int dt = 0; dt < 4; ++dt) {
    #pragma unroll
    for (int r = 0; r < 8; ++r) {
      int trow = qbase + mrow + r;
      float vv = o[dt][r] / lrun[r];
      size_t idx = ((size_t)(bb * 2048 + trow)) * 1024 + hh * 64 + dt * 16 + n16;
      Y[idx] = f2bf(vv);
    }
  }
}

// =====================================================================
// Kernel 3: out = Y(bf16) @ Wo + bo (fp32 out).  M=8192, N=1024, K=1024.
// Block tile 64 x 256, 8 wmma/wave/step.
// A tile is a raw bf16 copy -> use gfx1250 async global->LDS DMA
// (GLOBAL_LOAD_ASYNC_TO_LDS_B128, tracked by ASYNCcnt), bypassing VGPRs.
// =====================================================================
__global__ __launch_bounds__(256) void out_gemm_kernel(
    const unsigned short* __restrict__ Yb, const float* __restrict__ Wo,
    const float* __restrict__ bo, float* __restrict__ out)
{
  __shared__ unsigned short As[64][40];
  __shared__ unsigned int   Bs2[16][264];

  const int tid   = threadIdx.x;
  const int lane  = tid & 31;
  const int w     = tid >> 5;
  const int wm    = w & 1;
  const int wn    = w >> 1;
  const int mblk  = blockIdx.x * 64;
  const int nblk  = blockIdx.y * 256;
  const int nhalf = lane >> 4;
  const int n16   = lane & 15;

  // Per-thread A-tile slot (fixed across K-steps): LDS offset is the low
  // 32 bits of the generic pointer (LDS aperture keeps offset in addr[31:0]).
  const int arow = tid >> 2, ac8 = tid & 3;
  const unsigned int alds = (unsigned int)(size_t)&As[arow][ac8 * 8];

  f32x8 acc[2][4];
  #pragma unroll
  for (int bn = 0; bn < 4; ++bn) {
    float bv = bo[nblk + wn * 64 + bn * 16 + n16];
    #pragma unroll
    for (int am = 0; am < 2; ++am)
      #pragma unroll
      for (int r = 0; r < 8; ++r) acc[am][bn][r] = bv;
  }

  for (int kk = 0; kk < 1024; kk += 32) {
    __syncthreads();
    // stage A (already bf16): async DMA 16B per thread, global -> LDS
    {
      unsigned long long gsrc = (unsigned long long)(size_t)
          (Yb + (size_t)(mblk + arow) * 1024 + kk + ac8 * 8);
      asm volatile("global_load_async_to_lds_b128 %0, %1, off"
                   :: "v"(alds), "v"(gsrc)
                   : "memory");
    }
    // stage B pair-packed from Wo (needs f32->bf16 conversion)
    #pragma unroll
    for (int i = 0; i < 4; ++i) {
      int idx = tid + i * 256;
      int pr = idx >> 6, ch = idx & 63;
      const float* src = Wo + (size_t)(kk + 2 * pr) * 1024 + nblk + ch * 4;
      float4 r0 = *(const float4*)(src);
      float4 r1 = *(const float4*)(src + 1024);
      uint4 p;
      p.x = pk2bf(r0.x, r1.x);
      p.y = pk2bf(r0.y, r1.y);
      p.z = pk2bf(r0.z, r1.z);
      p.w = pk2bf(r0.w, r1.w);
      *(uint4*)&Bs2[pr][ch * 4] = p;
    }
    if (kk + 32 < 1024) {
      __builtin_prefetch(Yb + (size_t)(mblk + (tid & 63)) * 1024 + kk + 32, 0, 0);
      __builtin_prefetch(Wo + (size_t)(kk + 32 + (tid & 31)) * 1024 + nblk, 0, 0);
    }
    asm volatile("s_wait_asynccnt 0" ::: "memory");   // A DMA landed in LDS
    __syncthreads();

    Frag a[2], b[4];
    #pragma unroll
    for (int am = 0; am < 2; ++am) {
      int m = wm * 32 + am * 16 + n16;
      int khalf = nhalf * 8;
      #pragma unroll
      for (int v = 0; v < 4; ++v)
        a[am].u[v] = *(const unsigned int*)&As[m][khalf + 2 * v];
      #pragma unroll
      for (int v = 0; v < 4; ++v)
        a[am].u[4 + v] = *(const unsigned int*)&As[m][16 + khalf + 2 * v];
    }
    #pragma unroll
    for (int bn = 0; bn < 4; ++bn) {
      int n = wn * 64 + bn * 16 + n16;
      int kh2 = nhalf * 4;
      #pragma unroll
      for (int v = 0; v < 8; ++v)
        b[bn].u[v] = Bs2[(v >> 2) * 8 + kh2 + (v & 3)][n];
    }
    #pragma unroll
    for (int am = 0; am < 2; ++am)
      #pragma unroll
      for (int bn = 0; bn < 4; ++bn)
        acc[am][bn] = WMMA_BF16(a[am].v, b[bn].v, acc[am][bn]);
  }

  #pragma unroll
  for (int am = 0; am < 2; ++am)
    #pragma unroll
    for (int bn = 0; bn < 4; ++bn)
      #pragma unroll
      for (int r = 0; r < 8; ++r) {
        int gm = mblk + wm * 32 + am * 16 + r + 8 * nhalf;
        int gn = nblk + wn * 64 + bn * 16 + n16;
        out[(size_t)gm * 1024 + gn] = acc[am][bn][r];
      }
}

// =====================================================================
// Launcher
// =====================================================================
extern "C" void kernel_launch(void* const* d_in, const int* in_sizes, int n_in,
                              void* d_out, int out_size, void* d_ws, size_t ws_size,
                              hipStream_t stream) {
  const float* x    = (const float*)d_in[0];   // [4,2048,1024]
  const float* Wqkv = (const float*)d_in[1];   // [1024,3072]
  const float* bqkv = (const float*)d_in[2];   // [3072]
  const float* Wo   = (const float*)d_in[3];   // [1024,1024]
  const float* bo   = (const float*)d_in[4];   // [1024]
  float* out        = (float*)d_out;           // [4,2048,1024] fp32

  const size_t PER = (size_t)4 * 16 * 2048 * 64;   // 8,388,608 elements
  unsigned short* Q  = (unsigned short*)d_ws;      // [B,H,T,D] bf16
  unsigned short* K  = Q + PER;                    // [B,H,T,D] bf16
  unsigned short* Vt = K + PER;                    // [B,H,D,T] bf16
  unsigned short* Y  = Vt + PER;                   // [B*T, C]  bf16
  // total workspace use: 4 * 8,388,608 * 2B = 64 MB

  dim3 g1(128, 12);   // 8192/64 x 3072/256
  qkv_gemm_kernel<<<g1, 256, 0, stream>>>(x, Wqkv, bqkv, Q, K, Vt);

  attn_kernel<<<2048, 128, 0, stream>>>(Q, K, Vt, Y);   // 8192 waves, 1 per 16-query tile

  dim3 g3(128, 4);    // 8192/64 x 1024/256
  out_gemm_kernel<<<g3, 256, 0, stream>>>(Y, Wo, bo, out);
}
